// SwinTransformer3D_CNN_13022340842205
// MI455X (gfx1250) — matrix operand order repaired and checked by reading, not compiled, not run on hardware
//
#include <hip/hip_runtime.h>
#include <hip/hip_bf16.h>
#include <math.h>

// ---------------------------------------------------------------------------
// Swin-3D block on gfx1250 (MI455X), wave32 + WMMA f32<-f16 (16x16x32).
// Pipeline: cvt weights -> LN1+shift+partition -> qkv GEMM -> window attention
//           -> proj GEMM (+scatter residual into d_out) -> LN2 -> fc1+GELU
//           -> fc2 (+= into d_out).
// All activations staged in f16 (halves HBM traffic; f32 accumulate in WMMA).
// ---------------------------------------------------------------------------

typedef _Float16 half_t;
typedef __attribute__((ext_vector_type(16))) _Float16 v16h;
typedef __attribute__((ext_vector_type(8)))  _Float16 v8h;
typedef __attribute__((ext_vector_type(8)))  float    v8f;
typedef unsigned int u32;

#define TOK 200704   // 4*16*56*56 tokens == 2048 windows * 98
#define NEG_BIG (-3.0e38f)

__device__ __forceinline__ v16h ld16h(const half_t* p) {
  v8h lo = *(const v8h*)p;
  v8h hi = *(const v8h*)(p + 8);
  v16h r;
#pragma unroll
  for (int i = 0; i < 8; ++i) { r[i] = lo[i]; r[i + 8] = hi[i]; }
  return r;
}

// A-matrix fragment (16 rows x 32 k), tile row-major in LDS, stride in halves.
// Lane L (<16): row=L, K = 0..7,16..23 ; lane L(>=16): row=L-16, K = 8..15,24..31.
__device__ __forceinline__ v16h load_a_frag(const half_t* base, int stride, int lane) {
  int row = lane & 15;
  int kb  = (lane < 16) ? 0 : 8;
  const half_t* p = base + row * stride + kb;
  v8h lo = *(const v8h*)p;
  v8h hi = *(const v8h*)(p + 16);
  v16h r;
#pragma unroll
  for (int i = 0; i < 8; ++i) { r[i] = lo[i]; r[i + 8] = hi[i]; }
  return r;
}

// B-matrix fragment (32 k x 16 cols); LDS tile stored [col][k] (k contiguous).
// Lane L: col = L%16, K = (L<16 ? 0..7,16..23 : 8..15,24..31).
__device__ __forceinline__ v16h load_b_frag(const half_t* base, int stride, int lane) {
  int col = lane & 15;
  int kb  = (lane < 16) ? 0 : 8;
  const half_t* p = base + col * stride + kb;
  v8h lo = *(const v8h*)p;
  v8h hi = *(const v8h*)(p + 16);
  v16h r;
#pragma unroll
  for (int i = 0; i < 8; ++i) { r[i] = lo[i]; r[i + 8] = hi[i]; }
  return r;
}

__device__ __forceinline__ v8f wmma_f16(v16h a, v16h b, v8f c) {
  return __builtin_amdgcn_wmma_f32_16x16x32_f16(false, a, false, b, (short)0, c,
                                                false, false);
}

// windowed-token index -> original token index (includes cyclic shift (1,3,3))
__device__ __forceinline__ int win_to_orig(int t) {
  int w = t / 98, n = t - w * 98;
  int b = w >> 9, id = (w >> 6) & 7, ih = (w >> 3) & 7, iw = w & 7;
  int td = n / 49; int rem = n - td * 49;
  int th = rem / 7; int tw = rem - th * 7;
  int ds = (id * 2 + td + 1) & 15;
  int hs = (ih * 7 + th + 3) % 56;
  int ws = (iw * 7 + tw + 3) % 56;
  return ((b * 16 + ds) * 56 + hs) * 56 + ws;
}

// ---------------------------------------------------------------------------
__global__ __launch_bounds__(256) void cvt_f16_kernel(const float* __restrict__ src,
                                                      half_t* __restrict__ dst, int n) {
  int i = blockIdx.x * 256 + threadIdx.x;
  if (i < n) dst[i] = (half_t)src[i];
}

// Padded bias: bf[h][n*112+m] = bias_table[rel_idx[n*98+m]][h] for n,m < 98,
// NEG_BIG elsewhere -> softmax masking falls out of plain unconditional loads.
__global__ __launch_bounds__(256) void bias_expand_kernel(const float* __restrict__ bt,
                                                          const int* __restrict__ rel,
                                                          float* __restrict__ bf) {
  int i = blockIdx.x * 256 + threadIdx.x;          // over 4*112*112
  if (i < 4 * 112 * 112) {
    int h  = i / 12544;
    int rm = i - h * 12544;
    int n = rm / 112, m = rm - n * 112;
    float v = NEG_BIG;
    if (n < 98 && m < 98) v = bt[rel[n * 98 + m] * 4 + h];
    bf[i] = v;
  }
}

// One wave per token, 128 channels (4/lane), wave32 shuffle reductions.
// windowed=1: gather source through shift+partition map; output in windowed order.
__global__ __launch_bounds__(256) void layernorm_kernel(const float* __restrict__ src,
                                                        const float* __restrict__ g,
                                                        const float* __restrict__ b,
                                                        half_t* __restrict__ out,
                                                        int windowed) {
  int wave = threadIdx.x >> 5, lane = threadIdx.x & 31;
  int t = blockIdx.x * 8 + wave;               // grid = TOK/8 exactly
  int orig = windowed ? win_to_orig(t) : t;
  const float* row = src + (size_t)orig * 128;
  float v[4]; float s = 0.f, sq = 0.f;
#pragma unroll
  for (int j = 0; j < 4; ++j) { v[j] = row[lane + 32 * j]; s += v[j]; sq += v[j] * v[j]; }
#pragma unroll
  for (int m = 1; m < 32; m <<= 1) { s += __shfl_xor(s, m, 32); sq += __shfl_xor(sq, m, 32); }
  float mean = s * (1.f / 128.f);
  float var  = sq * (1.f / 128.f) - mean * mean;
  float rst  = rsqrtf(var + 1e-5f);
  half_t* orow = out + (size_t)t * 128;
#pragma unroll
  for (int j = 0; j < 4; ++j) {
    int c = lane + 32 * j;
    orow[c] = (half_t)((v[j] - mean) * rst * g[c] + b[c]);
  }
}

// ---------------------------------------------------------------------------
// Generic GEMM: Y[M,Nout] = A[M,K] * W[Nout,K]^T + bias, M-tile 128, N-tile 64.
// 8 waves: wave w computes 16-row strip, 4 col-tiles -> 4 WMMAs per k-chunk.
// EPI: 0 = f16 store (qkv)            1 = proj: scatter + residual -> f32 out
//      2 = GELU(exact) -> f16 store   3 = accumulate into f32 out (fc2)
template <int EPI>
__global__ __launch_bounds__(256) void gemm_kernel(const half_t* __restrict__ A, int lda,
                                                   const half_t* __restrict__ W, int K,
                                                   const float* __restrict__ bias,
                                                   half_t* __restrict__ outH, int ldo,
                                                   float* __restrict__ outF,
                                                   const float* __restrict__ resid) {
  __shared__ alignas(16) half_t As[128 * 32];
  __shared__ alignas(16) half_t Bs[64 * 32];
  int tid = threadIdx.x, wave = tid >> 5, lane = tid & 31;
  int rowBase = blockIdx.x * 128;
  int colBase = blockIdx.y * 64;
  v8f acc[4] = {};

  for (int k0 = 0; k0 < K; k0 += 32) {
#pragma unroll 2
    for (int i = tid; i < 2048; i += 256) {               // A: 128x32 halves
      int r = i >> 4, kc = (i & 15) * 2;
      ((u32*)As)[i] = *(const u32*)(A + (size_t)(rowBase + r) * lda + k0 + kc);
    }
#pragma unroll 2
    for (int i = tid; i < 1024; i += 256) {               // B: 64 cols x 32 k
      int n = i >> 4, kc = (i & 15) * 2;
      ((u32*)Bs)[i] = *(const u32*)(W + (size_t)(colBase + n) * K + k0 + kc);
    }
    if (k0 + 32 < K)                                      // gfx1250 global_prefetch_b8
      __builtin_prefetch(A + (size_t)(rowBase + (tid & 127)) * lda + k0 + 32, 0, 3);
    __syncthreads();
    // issue all fragment loads before the WMMA chain (overlap LDS with matrix pipe)
    v16h a = load_a_frag(As + wave * 16 * 32, 32, lane);
    v16h b[4];
#pragma unroll
    for (int ct = 0; ct < 4; ++ct) b[ct] = load_b_frag(Bs + ct * 16 * 32, 32, lane);
#pragma unroll
    for (int ct = 0; ct < 4; ++ct) acc[ct] = wmma_f16(a, b[ct], acc[ct]);
    __syncthreads();
  }

  int rhalf = (lane < 16) ? 0 : 8;
#pragma unroll
  for (int ct = 0; ct < 4; ++ct) {
    int col = colBase + ct * 16 + (lane & 15);
    float bv = bias[col];
#pragma unroll
    for (int r = 0; r < 8; ++r) {
      int grow = rowBase + wave * 16 + r + rhalf;
      float v = acc[ct][r] + bv;
      if (EPI == 0) {
        outH[(size_t)grow * ldo + col] = (half_t)v;
      } else if (EPI == 1) {  // window-reverse + reverse-shift scatter + residual
        int orig = win_to_orig(grow);
        outF[(size_t)orig * 128 + col] = resid[(size_t)orig * 128 + col] + v;
      } else if (EPI == 2) {  // exact GELU
        float gl = 0.5f * v * (1.f + erff(v * 0.70710678118654752f));
        outH[(size_t)grow * ldo + col] = (half_t)gl;
      } else {                // fc2: out += v (out already holds x1)
        outF[(size_t)grow * 128 + col] += v;
      }
    }
  }
}

// ---------------------------------------------------------------------------
// Attention: one block per (window, head). N=98 padded to 112, hd=32.
// Waves 0..6 each own one 16-row strip: 7 WMMAs for S=Q*K^T, register softmax
// with 16-lane shuffle reductions, P -> wave-private LDS, 4x2 WMMAs for P*V.
__global__ __launch_bounds__(256) void attn_kernel(const half_t* __restrict__ qkv,
                                                   const float* __restrict__ biasFull,
                                                   half_t* __restrict__ o) {
  __shared__ alignas(16) half_t Qs[112 * 32];
  __shared__ alignas(16) half_t Ks[112 * 32];
  __shared__ alignas(16) half_t Vst[32 * 128];      // transposed: [d][m]
  __shared__ alignas(16) half_t Ps[7 * 16 * 128];   // per-wave P strips
  int w = blockIdx.x >> 2, head = blockIdx.x & 3;
  int tid = threadIdx.x, wave = tid >> 5, lane = tid & 31;
  const half_t* base = qkv + (size_t)w * 98 * 384;
  int qoff = head * 32, koff = 128 + head * 32, voff = 256 + head * 32;

  for (int i = tid; i < 112 * 16; i += 256) {       // Q,K rows (zero-pad 98..111)
    int r = i >> 4, kc = (i & 15) * 2;
    u32 q = 0, k = 0;
    if (r < 98) {
      q = *(const u32*)(base + (size_t)r * 384 + qoff + kc);
      k = *(const u32*)(base + (size_t)r * 384 + koff + kc);
    }
    ((u32*)Qs)[i] = q;
    ((u32*)Ks)[i] = k;
  }
  for (int i = tid; i < 32 * 128; i += 256) {       // V transposed, zero-padded
    int d = i >> 7, m = i & 127;
    half_t v = (half_t)0.f;
    if (m < 98) v = base[(size_t)m * 384 + voff + d];
    Vst[i] = v;
  }
  __syncthreads();
  if (wave >= 7) return;                            // no further barriers

  int rt = wave;
  // preload A-frag + all 7 B-frags, then run the WMMA chain back-to-back
  v16h aq = load_a_frag(Qs + rt * 16 * 32, 32, lane);
  v16h bk[7];
#pragma unroll
  for (int ct = 0; ct < 7; ++ct) bk[ct] = load_b_frag(Ks + ct * 16 * 32, 32, lane);
  v8f s[7] = {};
#pragma unroll
  for (int ct = 0; ct < 7; ++ct) s[ct] = wmma_f16(aq, bk[ct], s[ct]);

  const float scale = 0.176776695296636893f;        // 32^-0.5
  int colL = lane & 15;
  int rhalf = (lane < 16) ? 0 : 8;
  half_t* Pw = Ps + wave * 16 * 128;
  const float* bh = biasFull + (size_t)head * 12544; // padded 112x112, NEG_BIG pad

#pragma unroll
  for (int r = 0; r < 8; ++r) {
    int row = rt * 16 + r + rhalf;
    const float* brow = bh + row * 112 + colL;
    float l[7];
    float mx = NEG_BIG;
#pragma unroll
    for (int ct = 0; ct < 7; ++ct) {                // unconditional, batched loads
      float lv = s[ct][r] * scale + brow[ct * 16];
      l[ct] = lv;
      mx = fmaxf(mx, lv);
    }
#pragma unroll
    for (int m = 1; m < 16; m <<= 1) mx = fmaxf(mx, __shfl_xor(mx, m, 32));
    float sum = 0.f;
#pragma unroll
    for (int ct = 0; ct < 7; ++ct) { float e = __expf(l[ct] - mx); l[ct] = e; sum += e; }
#pragma unroll
    for (int m = 1; m < 16; m <<= 1) sum += __shfl_xor(sum, m, 32);
    float inv = 1.f / sum;
    int lr = r + rhalf;
#pragma unroll
    for (int ct = 0; ct < 7; ++ct) Pw[lr * 128 + ct * 16 + colL] = (half_t)(l[ct] * inv);
    Pw[lr * 128 + 112 + colL] = (half_t)0.f;        // zero pad cols 112..127
  }

  // O = P x V : preload 4 A-frags and 8 B-frags, then chain the 8 WMMAs
  v16h ap[4];
#pragma unroll
  for (int kc2 = 0; kc2 < 4; ++kc2) ap[kc2] = load_a_frag(Pw + kc2 * 32, 128, lane);
  v16h bv[8];
#pragma unroll
  for (int kc2 = 0; kc2 < 4; ++kc2)
#pragma unroll
    for (int ct = 0; ct < 2; ++ct)
      bv[kc2 * 2 + ct] = load_b_frag(Vst + ct * 16 * 128 + kc2 * 32, 128, lane);
  v8f oacc[2] = {};
#pragma unroll
  for (int kc2 = 0; kc2 < 4; ++kc2)
#pragma unroll
    for (int ct = 0; ct < 2; ++ct)
      oacc[ct] = wmma_f16(ap[kc2], bv[kc2 * 2 + ct], oacc[ct]);

#pragma unroll
  for (int ct = 0; ct < 2; ++ct) {
#pragma unroll
    for (int r = 0; r < 8; ++r) {
      int row = rt * 16 + r + rhalf;
      if (row < 98)
        o[((size_t)w * 98 + row) * 128 + head * 32 + ct * 16 + colL] = (half_t)oacc[ct][r];
    }
  }
}

// ---------------------------------------------------------------------------
extern "C" void kernel_launch(void* const* d_in, const int* in_sizes, int n_in,
                              void* d_out, int out_size, void* d_ws, size_t ws_size,
                              hipStream_t stream) {
  (void)in_sizes; (void)n_in; (void)out_size; (void)ws_size;
  const float* x      = (const float*)d_in[0];
  const float* n1g    = (const float*)d_in[1];
  const float* n1b    = (const float*)d_in[2];
  const float* qkv_w  = (const float*)d_in[3];
  const float* qkv_b  = (const float*)d_in[4];
  const float* bt     = (const float*)d_in[5];
  const float* proj_w = (const float*)d_in[6];
  const float* proj_b = (const float*)d_in[7];
  const float* n2g    = (const float*)d_in[8];
  const float* n2b    = (const float*)d_in[9];
  const float* fc1_w  = (const float*)d_in[10];
  const float* fc1_b  = (const float*)d_in[11];
  const float* fc2_w  = (const float*)d_in[12];
  const float* fc2_b  = (const float*)d_in[13];
  const int*   rel    = (const int*)d_in[14];
  float* out = (float*)d_out;           // also serves as residual buffer x1

  char* ws = (char*)d_ws;
  size_t off = 0;
  auto take = [&](size_t bytes) {
    char* p = ws + off;
    off = (off + bytes + 255) & ~(size_t)255;
    return p;
  };
  half_t* qkvW     = (half_t*)take((size_t)3 * 128 * 128 * sizeof(half_t));
  half_t* projW    = (half_t*)take((size_t)128 * 128 * sizeof(half_t));
  half_t* fc1W     = (half_t*)take((size_t)512 * 128 * sizeof(half_t));
  half_t* fc2W     = (half_t*)take((size_t)128 * 512 * sizeof(half_t));
  float*  biasFull = (float*) take((size_t)4 * 112 * 112 * sizeof(float));
  half_t* bufB     = (half_t*)take((size_t)TOK * 128 * sizeof(half_t)); // hw / h2
  half_t* bufA     = (half_t*)take((size_t)TOK * 512 * sizeof(half_t)); // qkv / mlp1
  half_t* bufC     = (half_t*)take((size_t)TOK * 128 * sizeof(half_t)); // attn out

  cvt_f16_kernel<<<(49152 + 255) / 256, 256, 0, stream>>>(qkv_w, qkvW, 49152);
  cvt_f16_kernel<<<(16384 + 255) / 256, 256, 0, stream>>>(proj_w, projW, 16384);
  cvt_f16_kernel<<<(65536 + 255) / 256, 256, 0, stream>>>(fc1_w, fc1W, 65536);
  cvt_f16_kernel<<<(65536 + 255) / 256, 256, 0, stream>>>(fc2_w, fc2W, 65536);
  bias_expand_kernel<<<(4 * 112 * 112 + 255) / 256, 256, 0, stream>>>(bt, rel, biasFull);

  // LN1 + shift + window partition (windowed order, f16)
  layernorm_kernel<<<TOK / 8, 256, 0, stream>>>(x, n1g, n1b, bufB, 1);
  // qkv: (TOK,128) x (384,128)^T -> f16 (stride 384)
  gemm_kernel<0><<<dim3(TOK / 128, 6), 256, 0, stream>>>(bufB, 128, qkvW, 128, qkv_b,
                                                         bufA, 384, nullptr, nullptr);
  // windowed multi-head attention
  attn_kernel<<<2048 * 4, 256, 0, stream>>>(bufA, biasFull, bufC);
  // proj + window-reverse scatter + residual -> out (x1, f32)
  gemm_kernel<1><<<dim3(TOK / 128, 2), 256, 0, stream>>>(bufC, 128, projW, 128, proj_b,
                                                         nullptr, 0, out, x);
  // LN2 (natural order) -> f16
  layernorm_kernel<<<TOK / 8, 256, 0, stream>>>(out, n2g, n2b, bufB, 0);
  // fc1 + exact GELU -> f16 (stride 512)
  gemm_kernel<2><<<dim3(TOK / 128, 8), 256, 0, stream>>>(bufB, 128, fc1W, 128, fc1_b,
                                                         bufA, 512, nullptr, nullptr);
  // fc2, accumulate into out (out = x1 + mlp)
  gemm_kernel<3><<<dim3(TOK / 128, 2), 256, 0, stream>>>(bufA, 512, fc2W, 512, fc2_b,
                                                         nullptr, 0, out, nullptr);
}
